// DynamicGNN_9569187135766
// MI455X (gfx1250) — matrix operand order repaired and checked by reading, not compiled, not run on hardware
//
#include <hip/hip_runtime.h>
#include <hip/hip_bf16.h>
#include <math.h>

// Problem constants (from reference)
#define BQ 4
#define TQ 32
#define NQ 1024
#define EQ 8192
#define DINQ 8
#define HQ 128
#define H3 384
#define H2 64
#define CQ 2
#define MQ (BQ * TQ * NQ)   // 131072 nodes
#define BNQ (BQ * NQ)       // 4096 sequences
#define EMQ (BQ * TQ * EQ)  // 1048576 edges
#define EPSQ 1e-5f
#define KD 128              // all WMMA GEMMs have K = H = 128
#define KC (KD / 32)        // 4 k-chunks of 32

typedef __attribute__((ext_vector_type(16))) __bf16 v16bf;
typedef __attribute__((ext_vector_type(16))) unsigned short v16us;
typedef __attribute__((ext_vector_type(8))) unsigned short v8us;
typedef __attribute__((ext_vector_type(8))) float v8f;

__device__ __forceinline__ unsigned short f2bf(float f) {
  unsigned int u = __float_as_uint(f);
  unsigned int r = (u + 0x7FFFu + ((u >> 16) & 1u)) >> 16;  // RNE
  return (unsigned short)r;
}

// ---------------- utility kernels ----------------
__global__ void k_zero(float* p, long n) {
  long i = (long)blockIdx.x * blockDim.x + threadIdx.x;
  long stride = (long)gridDim.x * blockDim.x;
  for (; i < n; i += stride) p[i] = 0.0f;
}

// Pack weights (fp32) directly into WMMA-B fragment-native bf16 layout:
// Bp[tileN j][kchunk c][lane][elem i], elem i of lane L is B[k][n] with
//   k = c*32 + (L>>4)*16 + i ,  n = j*16 + (L&15)
// trans==0: W stored [K][N] row-major (e.g. Wr).
// trans==1: W stored [N][K] row-major (e.g. GRU Wih/Whh, effective B = W^T).
__global__ void k_pack_w(const float* Wf, unsigned short* Bp, int Ndim, int trans) {
  long total = (long)(Ndim / 16) * KC * 32 * 16;
  long idx = (long)blockIdx.x * blockDim.x + threadIdx.x;
  long stride = (long)gridDim.x * blockDim.x;
  for (; idx < total; idx += stride) {
    int i = (int)(idx & 15);
    int lane = (int)((idx >> 4) & 31);
    int c = (int)((idx >> 9) & (KC - 1));
    int j = (int)(idx >> 11);
    int k = c * 32 + ((lane >> 4) << 4) + i;
    int n = (j << 4) + (lane & 15);
    float v = trans ? Wf[(size_t)n * KD + k] : Wf[(size_t)k * Ndim + n];
    Bp[idx] = f2bf(v);
  }
}

// ---------------- graph preprocessing ----------------
__global__ void k_degree(const int* ei, float* deg) {
  long e = (long)blockIdx.x * blockDim.x + threadIdx.x;
  if (e >= EMQ) return;
  int ft = (int)(e / EQ);
  int k = (int)(e % EQ);
  int dst = ei[((long)ft * 2 + 1) * EQ + k];
  atomicAdd(&deg[(long)ft * NQ + dst], 1.0f);
}

__global__ void k_finalize_deg(const float* deg, float* dinv, float* selfn) {
  long i = (long)blockIdx.x * blockDim.x + threadIdx.x;
  if (i >= MQ) return;
  float d = deg[i] + 1.0f;  // self-loop
  dinv[i] = rsqrtf(d);
  selfn[i] = 1.0f / d;
}

// ---------------- GCN layer 0 dense part: hW = x @ W0 (K=8, plain VALU) ----
__global__ void k_gemm_x_w0(const float* x, const float* W0, float* hW) {
  long idx = (long)blockIdx.x * blockDim.x + threadIdx.x;
  if (idx >= (long)MQ * HQ) return;
  long n = idx >> 7;
  int f = (int)(idx & 127);
  float acc = 0.0f;
#pragma unroll
  for (int k = 0; k < DINQ; k++) acc += x[n * DINQ + k] * W0[k * HQ + f];
  hW[idx] = acc;
}

// ---------------- edge scatter: agg[dst] += snorm * hW[src] ----------------
__global__ void k_scatter(const int* ei, const float* hW, const float* dinv, float* agg) {
  long e = blockIdx.x;  // one block (128 threads) per edge
  int f = threadIdx.x;
  int ft = (int)(e / EQ);
  int k = (int)(e % EQ);
  int s = ei[((long)ft * 2 + 0) * EQ + k];
  int d = ei[((long)ft * 2 + 1) * EQ + k];
  long sg = (long)ft * NQ + s;
  long dg = (long)ft * NQ + d;
  float sn = dinv[sg] * dinv[dg];
  atomicAdd(&agg[dg * HQ + f], hW[sg * HQ + f] * sn);
}

// ---------------- GCN combine + LayerNorm + ReLU (+residual) ---------------
__global__ void k_combine(const float* agg, const float* hW, const float* selfn,
                          const float* bias, const float* gamma, const float* beta,
                          const float* resid, float* outp, unsigned short* out_bf) {
  long node = blockIdx.x;
  int f = threadIdx.x;
  __shared__ float red[HQ];
  float v = agg[node * HQ + f] + hW[node * HQ + f] * selfn[node] + bias[f];
  red[f] = v;
  __syncthreads();
  for (int s = 64; s > 0; s >>= 1) {
    if (f < s) red[f] += red[f + s];
    __syncthreads();
  }
  float mu = red[0] / (float)HQ;
  __syncthreads();
  float dvi = v - mu;
  red[f] = dvi * dvi;
  __syncthreads();
  for (int s = 64; s > 0; s >>= 1) {
    if (f < s) red[f] += red[f + s];
    __syncthreads();
  }
  float var = red[0] / (float)HQ;
  float o = dvi * rsqrtf(var + EPSQ) * gamma[f] + beta[f];
  o = fmaxf(o, 0.0f);
  if (resid) o += resid[node * HQ + f];
  outp[node * HQ + f] = o;
  out_bf[node * HQ + f] = f2bf(o);
}

// ---------------- WMMA bf16 GEMM: C[M,N] = A[M,128] @ B[128,N] --------------
// A: bf16 row-major; all 4 k-chunk fragments hoisted up front (2x b128 each).
// Bp: weights pre-packed fragment-native (k_pack_w); the WHOLE packed B
// (KD*Ndim bf16, <= 96 KB) is staged into LDS once per block and shared by all
// 8 waves. Inner loop uses a 4-slot rotating B-fragment buffer with prefetch
// depth 3, so 3 fragments are live at all times (distinct VGPR sets) and each
// ds_load has 3 WMMAs of latency cover -> no dscnt-0 stalls per WMMA.
template <int NT>
__global__ void __launch_bounds__(256) k_wmma_gemm(
    const unsigned short* __restrict__ A, const unsigned short* __restrict__ Bp,
    float* __restrict__ C, int Mdim, int Ndim) {
  extern __shared__ unsigned short ldsB[];  // KD*Ndim bf16, fragment-packed
  {
    int nvec = (KD * Ndim) >> 3;  // 16-byte chunks
    const v8us* src = (const v8us*)Bp;
    v8us* dst = (v8us*)ldsB;
    for (int i = threadIdx.x; i < nvec; i += blockDim.x) dst[i] = src[i];
  }
  __syncthreads();

  int wave = (int)(((long)blockIdx.x * blockDim.x + threadIdx.x) >> 5);
  int lane = threadIdx.x & 31;
  int groupsN = Ndim / (16 * NT);
  int tm = wave / groupsN;
  int tg = wave % groupsN;
  if (tm * 16 >= Mdim) return;
  int row0 = tm << 4;
  int ar = row0 + (lane & 15);
  int khalf = (lane >> 4) << 4;  // 0 or 16
  const unsigned short* Abase = A + (size_t)ar * KD;

  // hoist all A fragments (4 chunks x 2 contiguous b128 segments)
  v16us au[KC];
#pragma unroll
  for (int c = 0; c < KC; c++) {
    const unsigned short* ap = Abase + c * 32 + khalf;
    v8us a0 = *(const v8us*)(ap);       // K = khalf .. khalf+7
    v8us a1 = *(const v8us*)(ap + 8);   // K = khalf+8 .. khalf+15
    au[c] = __builtin_shufflevector(a0, a1, 0, 1, 2, 3, 4, 5, 6, 7, 8, 9, 10,
                                    11, 12, 13, 14, 15);
  }

  auto loadB = [&](int c, int j) -> v16us {
    const unsigned short* bp =
        ldsB + ((((size_t)(tg * NT + j)) * KC + c) * 32 + lane) * 16;
    v8us b0 = *(const v8us*)(bp);
    v8us b1 = *(const v8us*)(bp + 8);
    return __builtin_shufflevector(b0, b1, 0, 1, 2, 3, 4, 5, 6, 7, 8, 9, 10,
                                   11, 12, 13, 14, 15);
  };

  v8f acc[NT];
#pragma unroll
  for (int j = 0; j < NT; j++) acc[j] = v8f{};

  // 4-slot rotating fragment buffer, prefetch depth 3. Flattened index
  // idx = c*NT + j. Slot (idx+3)&3 is written while slots idx..idx+2 are live,
  // forcing distinct VGPR sets and keeping 3 fragments in flight.
  constexpr int TOT = KC * NT;
  v16us bb[4];
#pragma unroll
  for (int p = 0; p < 3 && p < TOT; p++) bb[p & 3] = loadB(p / NT, p % NT);
#pragma unroll
  for (int idx = 0; idx < TOT; idx++) {
    int pf = idx + 3;
    if (pf < TOT) bb[pf & 3] = loadB(pf / NT, pf % NT);
    int c = idx / NT;
    int j = idx % NT;
    acc[j] = __builtin_amdgcn_wmma_f32_16x16x32_bf16(
        false, __builtin_bit_cast(v16bf, au[c]), false,
        __builtin_bit_cast(v16bf, bb[idx & 3]), (short)0, acc[j], false, false);
  }

  int cr = row0 + ((lane >> 4) << 3);
#pragma unroll
  for (int j = 0; j < NT; j++) {
    int cc = (tg * NT + j) * 16 + (lane & 15);
#pragma unroll
    for (int i = 0; i < 8; i++) C[(size_t)(cr + i) * Ndim + cc] = acc[j][i];
  }
}

// ---------------- repack h[B,T,N,H] -> seq[T, B*N, H] (bf16) ----------------
__global__ void k_repack(const float* h, unsigned short* seq_bf) {
  long n = (long)MQ * HQ;
  long i = (long)blockIdx.x * blockDim.x + threadIdx.x;
  long stride = (long)gridDim.x * blockDim.x;
  for (; i < n; i += stride) {
    int f = (int)(i % HQ);
    long m = i / HQ;
    int nn = (int)(m % NQ);
    int ft = (int)(m / NQ);
    int b = ft / TQ, t = ft % TQ;
    seq_bf[((long)t * BNQ + (long)b * NQ + nn) * HQ + f] = f2bf(h[i]);
  }
}

// ---------------- GRU gate fusion ------------------------------------------
__global__ void k_gru_gate(const float* gi, const float* gh, const float* bih,
                           const float* bhh, float* hstate, unsigned short* hstate_bf,
                           float* outF, unsigned short* outBf) {
  long m = blockIdx.x;  // one block per sequence row
  int f = threadIdx.x;
  float ir = gi[m * H3 + f] + bih[f];
  float iz = gi[m * H3 + HQ + f] + bih[HQ + f];
  float in_ = gi[m * H3 + 2 * HQ + f] + bih[2 * HQ + f];
  float hr = gh[m * H3 + f] + bhh[f];
  float hz = gh[m * H3 + HQ + f] + bhh[HQ + f];
  float hn = gh[m * H3 + 2 * HQ + f] + bhh[2 * HQ + f];
  float r = 1.0f / (1.0f + __expf(-(ir + hr)));
  float z = 1.0f / (1.0f + __expf(-(iz + hz)));
  float nn = tanhf(in_ + r * hn);
  float hp = hstate[m * HQ + f];
  float hnew = (1.0f - z) * nn + z * hp;
  hstate[m * HQ + f] = hnew;
  unsigned short hb = f2bf(hnew);
  hstate_bf[m * HQ + f] = hb;
  outF[m * HQ + f] = hnew;
  outBf[m * HQ + f] = hb;
}

// ---------------- pooling + head -------------------------------------------
__global__ void k_frame(const float* seqF, const float* mask, float* frame) {
  int bt = blockIdx.x;  // b*T + t
  int b = bt / TQ, t = bt % TQ;
  int f = threadIdx.x;
  float acc = 0.0f, ms = 0.0f;
  for (int n = 0; n < NQ; n++) {
    float mv = mask[(long)bt * NQ + n];
    acc += seqF[((long)t * BNQ + (long)b * NQ + n) * HQ + f] * mv;
    ms += mv;
  }
  frame[(long)bt * HQ + f] = acc / fmaxf(ms, 1.0f);
}

__global__ void k_head(const float* frame, const float* Wc1, const float* bc1,
                       const float* Wc2, const float* bc2, float* outp) {
  int b = blockIdx.x;
  int tid = threadIdx.x;
  __shared__ float pooled[HQ];
  __shared__ float hid[H2];
  if (tid < HQ) {
    float a = 0.0f;
    for (int t = 0; t < TQ; t++) a += frame[((long)b * TQ + t) * HQ + tid];
    pooled[tid] = a / (float)TQ;
  }
  __syncthreads();
  if (tid < H2) {
    float a = bc1[tid];
    for (int f = 0; f < HQ; f++) a += pooled[f] * Wc1[f * H2 + tid];
    hid[tid] = fmaxf(a, 0.0f);
  }
  __syncthreads();
  if (tid < CQ) {
    float a = bc2[tid];
    for (int j = 0; j < H2; j++) a += hid[j] * Wc2[j * CQ + tid];
    outp[b * CQ + tid] = a;
  }
}

// ---------------- orchestration --------------------------------------------
extern "C" void kernel_launch(void* const* d_in, const int* in_sizes, int n_in,
                              void* d_out, int out_size, void* d_ws, size_t ws_size,
                              hipStream_t stream) {
  const float* x = (const float*)d_in[0];
  const float* mask = (const float*)d_in[1];
  const float* W0 = (const float*)d_in[2];
  const float* b0 = (const float*)d_in[3];
  const float* g0 = (const float*)d_in[4];
  const float* be0 = (const float*)d_in[5];
  const float* Wr = (const float*)d_in[6];
  const float* br = (const float*)d_in[7];
  const float* gr = (const float*)d_in[8];
  const float* ber = (const float*)d_in[9];
  const float* Wih = (const float*)d_in[10];
  const float* Whh = (const float*)d_in[11];
  const float* bih = (const float*)d_in[12];
  const float* bhh = (const float*)d_in[13];
  const float* Wc1 = (const float*)d_in[14];
  const float* bc1 = (const float*)d_in[15];
  const float* Wc2 = (const float*)d_in[16];
  const float* bc2 = (const float*)d_in[17];
  const int* ei = (const int*)d_in[18];
  float* outp = (float*)d_out;

  char* base = (char*)d_ws;
  size_t off = 0;
  auto alloc = [&](size_t bytes) -> char* {
    char* p = base + off;
    off = (off + bytes + 255) & ~(size_t)255;
    return p;
  };
  const long MH = (long)MQ * HQ;
  float* deg = (float*)alloc((size_t)MQ * 4);
  float* dinv = (float*)alloc((size_t)MQ * 4);
  float* selfn = (float*)alloc((size_t)MQ * 4);
  float* R1 = (float*)alloc((size_t)MH * 4);  // h (GCN act)  / GRU L2 out fp32
  float* R2 = (float*)alloc((size_t)MH * 4);  // hW           / GRU L1 out fp32
  float* R3 = (float*)alloc((size_t)MH * 4);  // agg          / GRU L1 out bf16 (reuse)
  unsigned short* R4 = (unsigned short*)alloc((size_t)MH * 2);  // h bf16 / seq1 bf16 / L2 out bf16
  float* gi = (float*)alloc((size_t)BNQ * H3 * 4);
  float* gh = (float*)alloc((size_t)BNQ * H3 * 4);
  float* hstate = (float*)alloc((size_t)BNQ * HQ * 4);
  unsigned short* hstate_bf = (unsigned short*)alloc((size_t)BNQ * HQ * 2);
  float* frame = (float*)alloc((size_t)BQ * TQ * HQ * 4);
  unsigned short* WrP = (unsigned short*)alloc((size_t)2 * KD * HQ * 2);
  unsigned short* WihP = (unsigned short*)alloc((size_t)2 * KD * H3 * 2);
  unsigned short* WhhP = (unsigned short*)alloc((size_t)2 * KD * H3 * 2);

  // --- pack weights into WMMA fragment-native bf16 layout (once per launch) ---
  for (int l = 0; l < 2; l++) {
    k_pack_w<<<64, 256, 0, stream>>>(Wr + (long)l * HQ * HQ, WrP + (long)l * KD * HQ, HQ, 0);
    k_pack_w<<<192, 256, 0, stream>>>(Wih + (long)l * H3 * HQ, WihP + (long)l * KD * H3, H3, 1);
    k_pack_w<<<192, 256, 0, stream>>>(Whh + (long)l * H3 * HQ, WhhP + (long)l * KD * H3, H3, 1);
  }

  // --- degrees / norms ---
  k_zero<<<1024, 256, 0, stream>>>(deg, MQ);
  k_degree<<<(EMQ + 255) / 256, 256, 0, stream>>>(ei, deg);
  k_finalize_deg<<<(MQ + 255) / 256, 256, 0, stream>>>(deg, dinv, selfn);

  // --- GCN layer 0 (K=8 dense via VALU, scatter via atomics) ---
  k_gemm_x_w0<<<(int)((MH + 255) / 256), 256, 0, stream>>>(x, W0, R2);
  k_zero<<<2048, 256, 0, stream>>>(R3, MH);
  k_scatter<<<EMQ, HQ, 0, stream>>>(ei, R2, dinv, R3);
  k_combine<<<MQ, HQ, 0, stream>>>(R3, R2, selfn, b0, g0, be0, nullptr, R1, R4);

  // --- GCN layers 1,2 (bf16 WMMA GEMM M=131072,N=128,K=128; NT=8) ---
  {
    int waves = (MQ / 16) * (HQ / (16 * 8));
    int blocks = (waves * 32 + 255) / 256;
    size_t ldsBytes = (size_t)KD * HQ * 2;  // 32 KB
    for (int l = 0; l < 2; l++) {
      k_wmma_gemm<8><<<blocks, 256, ldsBytes, stream>>>(R4, WrP + (long)l * KD * HQ, R2, MQ, HQ);
      k_zero<<<2048, 256, 0, stream>>>(R3, MH);
      k_scatter<<<EMQ, HQ, 0, stream>>>(ei, R2, dinv, R3);
      k_combine<<<MQ, HQ, 0, stream>>>(R3, R2, selfn, br + l * HQ, gr + l * HQ,
                                       ber + l * HQ, R1, R1, R4);
    }
  }

  // --- repack to [T, B*N, H] bf16 for GRU layer 1 ---
  k_repack<<<2048, 256, 0, stream>>>(R1, R4);

  // --- GRU layers (bf16 WMMA GEMMs M=4096,N=384,K=128 per step; NT=8) ---
  {
    int waves = (BNQ / 16) * (H3 / (16 * 8));
    int gblocks = (waves * 32 + 255) / 256;
    size_t ldsBytes = (size_t)KD * H3 * 2;  // 96 KB (fits 320 KB WGP LDS)
    for (int l = 0; l < 2; l++) {
      const unsigned short* in_bf = (l == 0) ? R4 : (const unsigned short*)R3;
      float* outF = (l == 0) ? R2 : R1;
      unsigned short* outBf = (l == 0) ? (unsigned short*)R3 : R4;
      k_zero<<<256, 256, 0, stream>>>(hstate, (long)BNQ * HQ);
      k_zero<<<256, 256, 0, stream>>>((float*)hstate_bf, (long)BNQ * HQ / 2);
      for (int t = 0; t < TQ; t++) {
        k_wmma_gemm<8><<<gblocks, 256, ldsBytes, stream>>>(
            in_bf + (long)t * BNQ * HQ, WihP + (long)l * KD * H3, gi, BNQ, H3);
        k_wmma_gemm<8><<<gblocks, 256, ldsBytes, stream>>>(
            hstate_bf, WhhP + (long)l * KD * H3, gh, BNQ, H3);
        k_gru_gate<<<BNQ, HQ, 0, stream>>>(gi, gh, bih + l * H3, bhh + l * H3, hstate,
                                           hstate_bf, outF + (long)t * BNQ * HQ,
                                           outBf + (long)t * BNQ * HQ);
      }
    }
  }

  // --- masked pooling + MLP head ---
  k_frame<<<BQ * TQ, HQ, 0, stream>>>(R1, mask, frame);
  k_head<<<BQ, HQ, 0, stream>>>(frame, Wc1, bc1, Wc2, bc2, outp);
}
